// DecoderRNN_7387343749725
// MI455X (gfx1250) — matrix operand order repaired and checked by reading, not compiled, not run on hardware
//
#include <hip/hip_runtime.h>
#include <stdint.h>

// ---------------------------------------------------------------------------
// Problem constants (DecoderRNN reference)
// ---------------------------------------------------------------------------
#define BB 128
#define TT 64
#define EE 512
#define HH 512
#define AA 512
#define VV 10000

typedef __attribute__((ext_vector_type(16))) __bf16 v16bf;
typedef __attribute__((ext_vector_type(8)))  float  v8f;

struct alignas(16) BF8   { uint16_t h[8];  };   // half of an A fragment
struct alignas(32) BF16x { uint16_t h[16]; };   // 16 bf16 values (unused)

__device__ __forceinline__ uint16_t f2bf(float f) {
    union { float f; uint32_t u; } x; x.f = f;
    uint32_t u = x.u;
    u += 0x7FFFu + ((u >> 16) & 1u);            // round-to-nearest-even
    return (uint16_t)(u >> 16);
}

// A-matrix 16x32 bf16 fragment (ISA 7.12.2): lanes 0-15 -> K 0..7 & 16..23,
// lanes 16-31 -> K 8..15 & 24..31; M = lane & 15.
__device__ __forceinline__ v16bf load_a(const uint16_t* row, int k0, int kbase) {
    union { v16bf v; BF8 p[2]; } u;
    u.p[0] = *(const BF8*)(row + k0 + kbase);
    u.p[1] = *(const BF8*)(row + k0 + kbase + 16);
    return u.v;
}

// B-matrix 32x16 bf16 fragment: N = lane & 15, lanes 0-15 hold K 0..7 & 16..23,
// lanes 16-31 hold K 8..15 & 24..31 (same K order as A) -> two 16-byte loads per lane.
__device__ __forceinline__ v16bf load_b(const uint16_t* row, int k0, int khalf) {
    union { v16bf v; BF8 p[2]; } u;
    u.p[0] = *(const BF8*)(row + k0 + (khalf >> 1));
    u.p[1] = *(const BF8*)(row + k0 + (khalf >> 1) + 16);
    return u.v;
}

// ---------------------------------------------------------------------------
// WMMA GEMM:  C[M,N] = concatK(A0,A1) * W^T + bias
//   One wave per workgroup; wave tile = 64(M) x 16(N); tile coords come only
//   from blockIdx (SGPRs) -> no divergent control flow around v_wmma.
//   grid = (N/16, M/64); requires M % 64 == 0, N % 16 == 0, Ksplit % 32 == 0.
//   K axis split at Ksplit between A0 / A1 (both bf16 row-major) so the
//   concat(x,h)-style inputs never get materialized.
//   OUT_MODE: 0 = f32 out, 1 = bf16 out, 2 = f32 out with length masking
//   (row -> (t = row/128, b = row%128); zero if t >= lens[b]).
// ---------------------------------------------------------------------------
template <int OUT_MODE>
__global__ __launch_bounds__(32)
void wmma_gemm(const uint16_t* __restrict__ A0, int lda0,
               const uint16_t* __restrict__ A1, int lda1,
               const uint16_t* __restrict__ W,
               const float*    __restrict__ bias,
               void*           __restrict__ Cout,
               int N, int K, int Ksplit,
               const int*      __restrict__ lens)
{
    const int lane  = threadIdx.x;              // 0..31
    const int n0    = blockIdx.x << 4;
    const int m0    = blockIdx.y << 6;

    const int arow  = lane & 15;
    const int kbase = (lane >> 4) << 3;         // 0 or 8   (A fragment)
    const int khalf = (lane >> 4) << 4;         // 0 or 16  (B fragment)
    const int ncol  = lane & 15;

    const uint16_t* wrow = W + (size_t)(n0 + ncol) * K;

    v8f zero = {0.f,0.f,0.f,0.f,0.f,0.f,0.f,0.f};
    v8f acc[4] = {zero, zero, zero, zero};

    // --- K segment sourced from A0 ---
    {
        const uint16_t* ar = A0 + (size_t)(m0 + arow) * lda0;
        for (int k0 = 0; k0 < Ksplit; k0 += 32) {
            v16bf b = load_b(wrow, k0, khalf);
#pragma unroll
            for (int i = 0; i < 4; ++i) {
                v16bf a = load_a(ar + (size_t)(i << 4) * lda0, k0, kbase);
                acc[i] = __builtin_amdgcn_wmma_f32_16x16x32_bf16(
                    false, a, false, b, (short)0, acc[i], false, false);
            }
        }
    }
    // --- K segment sourced from A1 ---
    {
        const uint16_t* ar = A1 + (size_t)(m0 + arow) * lda1;
        for (int k0 = Ksplit; k0 < K; k0 += 32) {
            v16bf b = load_b(wrow, k0, khalf);
#pragma unroll
            for (int i = 0; i < 4; ++i) {
                v16bf a = load_a(ar + (size_t)(i << 4) * lda1, k0 - Ksplit, kbase);
                acc[i] = __builtin_amdgcn_wmma_f32_16x16x32_bf16(
                    false, a, false, b, (short)0, acc[i], false, false);
            }
        }
    }

    // --- epilogue: C/D layout row = r + 8*(lane>=16), col = lane&15 ---
    const int roff = (lane >> 4) << 3;
    const int col  = n0 + ncol;
    const float bv = bias[col];
#pragma unroll
    for (int i = 0; i < 4; ++i) {
#pragma unroll
        for (int r = 0; r < 8; ++r) {
            const int row = m0 + (i << 4) + r + roff;
            float v = acc[i][r] + bv;
            if (OUT_MODE == 2)
                v = ((row >> 7) < lens[row & 127]) ? v : 0.f;
            if (OUT_MODE == 1)
                ((uint16_t*)Cout)[(size_t)row * N + col] = f2bf(v);
            else
                ((float*)Cout)[(size_t)row * N + col] = v;
        }
    }
}

// ---------------------------------------------------------------------------
// Pointwise / setup kernels
// ---------------------------------------------------------------------------
__global__ void k_f32_to_bf16(const float* __restrict__ s, uint16_t* __restrict__ d, int n) {
    int i = blockIdx.x * blockDim.x + threadIdx.x;
    if (i < n) d[i] = f2bf(s[i]);
}

__global__ void k_build_wcat(const float* __restrict__ Wih, const float* __restrict__ Whh,
                             uint16_t* __restrict__ Wcat) {
    int i = blockIdx.x * blockDim.x + threadIdx.x;           // 2048*1024
    if (i >= 2048 * 1024) return;
    int k = i & 1023, n = i >> 10;
    float v = (k < 512) ? Wih[n * 512 + k] : Whh[n * 512 + (k - 512)];
    Wcat[i] = f2bf(v);
}

__global__ void k_bias_sum(const float* a, const float* b, float* d, int n) {
    int i = blockIdx.x * blockDim.x + threadIdx.x;
    if (i < n) d[i] = a[i] + b[i];
}

__global__ void k_build_xs(const float* __restrict__ feat, const int* __restrict__ cap,
                           const float* __restrict__ emb, uint16_t* __restrict__ xs) {
    int i = blockIdx.x * blockDim.x + threadIdx.x;           // B*T*E
    if (i >= BB * TT * EE) return;
    int e = i & (EE - 1), t = (i >> 9) & (TT - 1), b = i >> 15;
    float v;
    if (t == 0) v = feat[b * EE + e];
    else        v = emb[(size_t)cap[b * (TT - 1) + (t - 1)] * EE + e];
    xs[i] = f2bf(v);
}

__global__ void k_zero_state(uint16_t* hbf, float* c, int n) {
    int i = blockIdx.x * blockDim.x + threadIdx.x;
    if (i < n) { hbf[i] = 0; c[i] = 0.f; }
}

// softmax over A=512 then multiply by cnn_features -> attended (bf16)
__global__ void k_softmax_attend(const float* __restrict__ sc, const float* __restrict__ cnn,
                                 uint16_t* __restrict__ att) {
    __shared__ float red[256];
    int b = blockIdx.x, tid = threadIdx.x;
    const float* s = sc + (size_t)b * AA;
    float m = -3.4e38f;
    for (int i = tid; i < AA; i += 256) m = fmaxf(m, s[i]);
    red[tid] = m; __syncthreads();
    for (int o = 128; o > 0; o >>= 1) { if (tid < o) red[tid] = fmaxf(red[tid], red[tid + o]); __syncthreads(); }
    m = red[0]; __syncthreads();
    float sum = 0.f;
    for (int i = tid; i < AA; i += 256) sum += __expf(s[i] - m);
    red[tid] = sum; __syncthreads();
    for (int o = 128; o > 0; o >>= 1) { if (tid < o) red[tid] += red[tid + o]; __syncthreads(); }
    float inv = 1.f / red[0];
    for (int i = tid; i < AA; i += 256) {
        float aw = __expf(s[i] - m) * inv;
        att[(size_t)b * AA + i] = f2bf(cnn[(size_t)b * AA + i] * aw);
    }
}

// LSTM gate nonlinearities + state update + length masking.
__global__ void k_lstm_update(const float* __restrict__ g, float* __restrict__ c,
                              uint16_t* __restrict__ hbf, uint16_t* __restrict__ hid,
                              const int* __restrict__ lens, int t) {
    int i = blockIdx.x * blockDim.x + threadIdx.x;           // B*H
    if (i >= BB * HH) return;
    int b = i >> 9, j = i & (HH - 1);
    const float* gr = g + (size_t)b * 4 * HH;
    float gi = gr[j], gf = gr[HH + j], gg = gr[2 * HH + j], go = gr[3 * HH + j];
    float si = 1.f / (1.f + __expf(-gi));
    float sf = 1.f / (1.f + __expf(-gf));
    float tg = tanhf(gg);
    float so = 1.f / (1.f + __expf(-go));
    float c2 = sf * c[i] + si * tg;
    float h2 = so * tanhf(c2);
    bool active = t < lens[b];
    if (active) { c[i] = c2; hbf[i] = f2bf(h2); }
    hid[(size_t)t * BB * HH + i] = active ? f2bf(h2) : (uint16_t)0;
}

// ---------------------------------------------------------------------------
// Host-side orchestration (graph-capture safe: only stream launches)
// ---------------------------------------------------------------------------
extern "C" void kernel_launch(void* const* d_in, const int* in_sizes, int n_in,
                              void* d_out, int out_size, void* d_ws, size_t ws_size,
                              hipStream_t stream) {
    (void)in_sizes; (void)n_in; (void)out_size; (void)ws_size;
    const float* features = (const float*)d_in[0];
    const int*   captions = (const int*)d_in[1];
    const int*   lengths  = (const int*)d_in[2];
    const float* cnn      = (const float*)d_in[3];
    const float* emb_W    = (const float*)d_in[4];
    const float* W_ih     = (const float*)d_in[5];
    const float* W_hh     = (const float*)d_in[6];
    const float* b_ih     = (const float*)d_in[7];
    const float* b_hh     = (const float*)d_in[8];
    const float* att_W    = (const float*)d_in[9];
    const float* att_b    = (const float*)d_in[10];
    const float* attd_W   = (const float*)d_in[11];
    const float* attd_b   = (const float*)d_in[12];
    const float* out_W    = (const float*)d_in[13];
    const float* out_b    = (const float*)d_in[14];
    float* out = (float*)d_out;

    char* ws = (char*)d_ws;                                  // 256B-aligned slabs
    uint16_t* xs_bf   = (uint16_t*)(ws + 0);                 //  8,388,608
    uint16_t* attW_bf = (uint16_t*)(ws + 8388608);           //  1,048,576
    uint16_t* atdW_bf = (uint16_t*)(ws + 9437184);           //  1,048,576
    uint16_t* wcat_bf = (uint16_t*)(ws + 10485760);          //  4,194,304
    uint16_t* outW_bf = (uint16_t*)(ws + 14680064);          // 10,240,000
    float*    bsum    = (float*)   (ws + 24920064);          //      8,192
    uint16_t* h_bf    = (uint16_t*)(ws + 24928256);          //    131,072
    float*    c_st    = (float*)   (ws + 25059328);          //    262,144
    float*    scores  = (float*)   (ws + 25321472);          //    262,144
    uint16_t* att_bf  = (uint16_t*)(ws + 25583616);          //    131,072
    uint16_t* x2_bf   = (uint16_t*)(ws + 25714688);          //    131,072
    float*    gates   = (float*)   (ws + 25845760);          //  1,048,576
    uint16_t* hid_bf  = (uint16_t*)(ws + 26894336);          //  8,388,608

    dim3 blk(256);
    auto cdiv = [](int a, int b) { return (a + b - 1) / b; };

    // --- one-time setup ---
    k_f32_to_bf16<<<cdiv(AA * (HH + EE), 256), blk, 0, stream>>>(att_W,  attW_bf, AA * (HH + EE));
    k_f32_to_bf16<<<cdiv(EE * (AA + EE), 256), blk, 0, stream>>>(attd_W, atdW_bf, EE * (AA + EE));
    k_f32_to_bf16<<<cdiv(VV * HH, 256),        blk, 0, stream>>>(out_W,  outW_bf, VV * HH);
    k_build_wcat <<<cdiv(2048 * 1024, 256),    blk, 0, stream>>>(W_ih, W_hh, wcat_bf);
    k_bias_sum   <<<cdiv(4 * HH, 256),         blk, 0, stream>>>(b_ih, b_hh, bsum, 4 * HH);
    k_build_xs   <<<cdiv(BB * TT * EE, 256),   blk, 0, stream>>>(features, captions, emb_W, xs_bf);
    k_zero_state <<<cdiv(BB * HH, 256),        blk, 0, stream>>>(h_bf, c_st, BB * HH);

    dim3 wblk(32);                       // one wave per workgroup
    // OUT_MODE 0: f32
    auto gemm_f32 = [&](const uint16_t* A0, int lda0, const uint16_t* A1, int lda1,
                        const uint16_t* W, const float* bias, float* C,
                        int M, int N, int K, int Ksplit) {
        dim3 grid(N / 16, M / 64);
        wmma_gemm<0><<<grid, wblk, 0, stream>>>(A0, lda0, A1, lda1, W, bias, C,
                                                N, K, Ksplit, nullptr);
    };
    // OUT_MODE 1: bf16
    auto gemm_bf = [&](const uint16_t* A0, int lda0, const uint16_t* A1, int lda1,
                       const uint16_t* W, const float* bias, uint16_t* C,
                       int M, int N, int K, int Ksplit) {
        dim3 grid(N / 16, M / 64);
        wmma_gemm<1><<<grid, wblk, 0, stream>>>(A0, lda0, A1, lda1, W, bias, C,
                                                N, K, Ksplit, nullptr);
    };

    // --- step 0: plain LSTM on raw features, h=c=0 ---
    gemm_f32(xs_bf, TT * EE, h_bf, HH, wcat_bf, bsum, gates,
             BB, 4 * HH, EE + HH, EE);
    k_lstm_update<<<cdiv(BB * HH, 256), blk, 0, stream>>>(gates, c_st, h_bf, hid_bf, lengths, 0);

    // --- steps 1..T-1: attention -> x2 -> LSTM ---
    for (int t = 1; t < TT; ++t) {
        const uint16_t* xt = xs_bf + (size_t)t * EE;         // row b at b*T*E + t*E
        // scores = concat(x_t, h) @ att_W^T + att_b
        gemm_f32(xt, TT * EE, h_bf, HH, attW_bf, att_b, scores,
                 BB, AA, EE + HH, EE);
        k_softmax_attend<<<BB, blk, 0, stream>>>(scores, cnn, att_bf);
        // x2 = concat(x_t, attended) @ attd_W^T + attd_b  (bf16 directly)
        gemm_bf(xt, TT * EE, att_bf, AA, atdW_bf, attd_b, x2_bf,
                BB, EE, EE + AA, EE);
        // gates = concat(x2, h) @ [W_ih|W_hh]^T + (b_ih + b_hh)
        gemm_f32(x2_bf, EE, h_bf, HH, wcat_bf, bsum, gates,
                 BB, 4 * HH, EE + HH, EE);
        k_lstm_update<<<cdiv(BB * HH, 256), blk, 0, stream>>>(gates, c_st, h_bf, hid_bf, lengths, t);
    }

    // --- logits = hidden @ out_W^T + out_b, masked by lengths ---
    {
        dim3 grid(VV / 16, (TT * BB) / 64);
        wmma_gemm<2><<<grid, wblk, 0, stream>>>(hid_bf, HH, hid_bf, HH, outW_bf,
                                                out_b, out, VV, HH, HH, lengths);
    }
}
